// DirectG2PNEFS_45629732553293
// MI455X (gfx1250) — compile-verified, matrix operand order checked by reading, not run on hardware
//
#include <hip/hip_runtime.h>

typedef __bf16 bf16;
typedef bf16  v16bf __attribute__((ext_vector_type(16)));
typedef bf16  v8bf  __attribute__((ext_vector_type(8)));
typedef float v8f   __attribute__((ext_vector_type(8)));

#define DEV_INLINE __device__ __forceinline__
#define LDP 40  // padded LDS K-stride (80B rows: 16B aligned, conflict-skewed)

// ---- CDNA5 async global->LDS path (probe via __has_builtin) ----
#if defined(__gfx1250__) && __has_builtin(__builtin_amdgcn_global_load_async_to_lds_b128)
#define HAS_ASYNC_LDS 1
#else
#define HAS_ASYNC_LDS 0
#endif

#if HAS_ASYNC_LDS
typedef int v4i __attribute__((ext_vector_type(4)));
typedef __attribute__((address_space(1))) v4i* gv4p;  // global int4*
typedef __attribute__((address_space(3))) v4i* lv4p;  // LDS int4*
#endif

// Stage one 16B chunk into LDS: async DMA on CDNA5, else VGPR round-trip.
DEV_INLINE void stage16(const bf16* g, bf16* l) {
#if HAS_ASYNC_LDS
  __builtin_amdgcn_global_load_async_to_lds_b128((gv4p)g, (lv4p)l, 0, 0);
#else
  *(uint4*)l = *(const uint4*)g;
#endif
}

// Wait until our wave's ASYNCcnt <= N (no-op on the sync fallback path).
template <int N>
DEV_INLINE void wait_async() {
#if HAS_ASYNC_LDS
#if __has_builtin(__builtin_amdgcn_s_wait_asynccnt)
  __builtin_amdgcn_s_wait_asynccnt((unsigned short)N);
#else
  asm volatile("s_wait_asynccnt %0" ::"n"(N) : "memory");
#endif
#endif
}

DEV_INLINE v8f wmma_bf16(v16bf a, v16bf b, v8f c) {
  return __builtin_amdgcn_wmma_f32_16x16x32_bf16(false, a, false, b, (short)0, c,
                                                 false, false);
}

// Load a 16x32 bf16 fragment from an LDS tile laid out [row][k] with stride ld.
// Lanes 0-15: row=lane, K {0..7,16..23}; lanes 16-31: row=lane-16, K {8..15,24..31}.
DEV_INLINE v16bf load_frag(const bf16* base, int ld, int lane) {
  const int r  = lane & 15;
  const int kg = (lane >> 4) << 3;
  v8bf lo = *reinterpret_cast<const v8bf*>(base + r * ld + kg);
  v8bf hi = *reinterpret_cast<const v8bf*>(base + r * ld + 16 + kg);
  return __builtin_shufflevector(lo, hi, 0, 1, 2, 3, 4, 5, 6, 7, 8, 9, 10, 11,
                                 12, 13, 14, 15);
}

DEV_INLINE float sigm(float x) { return 1.f / (1.f + __expf(-x)); }

// ---------------------------------------------------------------- utilities
__global__ void cvt_f32_bf16(const float* __restrict__ s, bf16* __restrict__ d,
                             int n) {
  int i = blockIdx.x * 256 + threadIdx.x;
  if (i < n) d[i] = (bf16)s[i];
}

__global__ void embed_gather(const int* __restrict__ tok,
                             const float* __restrict__ emb,
                             bf16* __restrict__ x, int n /* BT*256 */) {
  int i = blockIdx.x * 256 + threadIdx.x;
  if (i < n) {
    int t = tok[i >> 8];
    x[i] = (bf16)emb[t * 256 + (i & 255)];
  }
}

__global__ void init_dec_inp(const bf16* __restrict__ enc,
                             bf16* __restrict__ inp, int T) {
  int i = blockIdx.x * 256 + threadIdx.x;  // over B*512
  int b = i >> 9, j = i & 511;
  inp[i] = enc[(size_t)b * T * 512 + j];
}

// -------------------------------------------------- generic WMMA GEMM + bias
// out[m, n] = sum_k A[m,k] * W[n,k] + bias[n]
// A: M x K bf16 row-major (row stride K); W: N x K bf16 row-major.
// grid = (N/128, M/128), block = 256 (8 waves), tile 128x128x32,
// double-buffered LDS with async global->LDS staging.
__global__ __launch_bounds__(256) void gemm_bias(
    const bf16* __restrict__ A, const bf16* __restrict__ W,
    const float* __restrict__ bias, float* __restrict__ out, int K,
    long out_stride) {
  __shared__ bf16 As[2][128][LDP];
  __shared__ bf16 Bs[2][128][LDP];
  const int tid = threadIdx.x;
  const int lane = tid & 31, wave = tid >> 5;
  const int wrow = wave >> 1, wcol = wave & 1;  // 4x2 wave grid
  const long bm = (long)blockIdx.y * 128;
  const int bn = blockIdx.x * 128;

  v8f acc[2][4];
#pragma unroll
  for (int i = 0; i < 2; ++i)
#pragma unroll
    for (int j = 0; j < 4; ++j)
#pragma unroll
      for (int e = 0; e < 8; ++e) acc[i][j][e] = 0.f;

  // 4 async-load instructions per stage (2 chunks A + 2 chunks B per thread)
  auto stage = [&](int buf, int kk) {
#pragma unroll
    for (int it = 0; it < 2; ++it) {
      int c = it * 256 + tid;
      int row = c >> 2, kq = c & 3;
      stage16(&A[(size_t)(bm + row) * K + kk + kq * 8], &As[buf][row][kq * 8]);
      stage16(&W[(size_t)(bn + row) * K + kk + kq * 8], &Bs[buf][row][kq * 8]);
    }
  };

  const int nk = K >> 5;
  stage(0, 0);
  for (int ks = 0; ks < nk; ++ks) {
    const int cur = ks & 1;
    if (ks + 1 < nk) {
      stage(cur ^ 1, (ks + 1) << 5);
      wait_async<4>();  // the 4 newest are next-tile; current tile is resident
    } else {
      wait_async<0>();
    }
    __syncthreads();
    v16bf af[2], bw[4];
#pragma unroll
    for (int i = 0; i < 2; ++i)
      af[i] = load_frag(&As[cur][(2 * wrow + i) * 16][0], LDP, lane);
#pragma unroll
    for (int j = 0; j < 4; ++j)
      bw[j] = load_frag(&Bs[cur][(4 * wcol + j) * 16][0], LDP, lane);
#pragma unroll
    for (int i = 0; i < 2; ++i)
#pragma unroll
      for (int j = 0; j < 4; ++j) acc[i][j] = wmma_bf16(af[i], bw[j], acc[i][j]);
    __syncthreads();
  }

#pragma unroll
  for (int i = 0; i < 2; ++i)
#pragma unroll
    for (int j = 0; j < 4; ++j) {
      int col = bn + (4 * wcol + j) * 16 + (lane & 15);
      float bv = bias[col];
#pragma unroll
      for (int r = 0; r < 8; ++r) {
        long m = bm + (2 * wrow + i) * 16 + r + 8 * (lane >> 4);
        out[m * out_stride + col] = acc[i][j][r] + bv;
      }
    }
}

// --------------------------------------------------------- fused LSTM step
// gates[b, g*256+j] = (x @ wih^T)[b,.] + (h_in @ whh^T)[b,.] + bias / xg_t
// then c/h update, all inside one workgroup per 64(batch) x 64(hidden) block
// so i,f,g,o for an element are workgroup-local. grid = (256/64, 256/64).
// Encoder: x==null, xg!=null (bias folded into xg). Decoder: xg==null.
__global__ __launch_bounds__(256) void lstm_step(
    const bf16* __restrict__ x, int in_dim, const bf16* __restrict__ wih,
    const bf16* __restrict__ h_in, const bf16* __restrict__ whh,
    const float* __restrict__ bias, const float* __restrict__ xg, int t,
    float* __restrict__ c, bf16* __restrict__ h_out,
    bf16* __restrict__ hstream, long hs_stride, long hs_off) {
  __shared__ bf16 As[2][64][LDP];
  __shared__ bf16 Bs[2][256][LDP];  // 4 gates x 64 rows
  const int tid = threadIdx.x;
  const int lane = tid & 31, wave = tid >> 5;
  const int mt = wave & 3;          // 16-row tile within 64 batch rows
  const int nt0 = (wave >> 2) * 2;  // first of two 16-col tiles
  const int bm = blockIdx.y * 64;   // batch block
  const int cj = blockIdx.x * 64;   // hidden-column block

  v8f acc[4][2];
#pragma unroll
  for (int g = 0; g < 4; ++g)
#pragma unroll
    for (int j = 0; j < 2; ++j)
#pragma unroll
      for (int e = 0; e < 8; ++e) acc[g][j][e] = 0.f;

  const int nk0 = x ? (in_dim >> 5) : 0;
  const int nkt = nk0 + 8;  // + 256/32 k-steps over h @ whh

  // 5 async-load instructions per stage (1 A chunk + 4 W chunks per thread)
  auto stage = [&](int buf, int ks) {
    const bool ph0 = ks < nk0;
    const bf16* Asrc = ph0 ? x : h_in;
    const bf16* Wsrc = ph0 ? wih : whh;
    const int aK = ph0 ? in_dim : 256;
    const int kk = (ph0 ? ks : ks - nk0) << 5;
    {
      int row = tid >> 2, kq = tid & 3;  // 64x32 A tile
      stage16(&Asrc[(size_t)(bm + row) * aK + kk + kq * 8],
              &As[buf][row][kq * 8]);
    }
#pragma unroll
    for (int it = 0; it < 4; ++it) {  // 256x32 W tile
      int cc = it * 256 + tid;
      int rr = cc >> 2, kq = cc & 3;
      int g = rr >> 6, nl = rr & 63;
      stage16(&Wsrc[(size_t)(g * 256 + cj + nl) * aK + kk + kq * 8],
              &Bs[buf][rr][kq * 8]);
    }
  };

  stage(0, 0);
  for (int ks = 0; ks < nkt; ++ks) {
    const int cur = ks & 1;
    if (ks + 1 < nkt) {
      stage(cur ^ 1, ks + 1);
      wait_async<5>();
    } else {
      wait_async<0>();
    }
    __syncthreads();
    v16bf af = load_frag(&As[cur][mt * 16][0], LDP, lane);
#pragma unroll
    for (int g = 0; g < 4; ++g)
#pragma unroll
      for (int j = 0; j < 2; ++j) {
        v16bf bw = load_frag(&Bs[cur][g * 64 + (nt0 + j) * 16][0], LDP, lane);
        acc[g][j] = wmma_bf16(af, bw, acc[g][j]);
      }
    __syncthreads();
  }

  // fused cell update: i,f,g,o all live in this wave's accumulators
#pragma unroll
  for (int j = 0; j < 2; ++j) {
    int J = cj + (nt0 + j) * 16 + (lane & 15);
#pragma unroll
    for (int r = 0; r < 8; ++r) {
      int m = bm + mt * 16 + r + 8 * (lane >> 4);
      float g0 = acc[0][j][r], g1 = acc[1][j][r];
      float g2 = acc[2][j][r], g3 = acc[3][j][r];
      if (bias) {
        g0 += bias[J];       g1 += bias[256 + J];
        g2 += bias[512 + J]; g3 += bias[768 + J];
      }
      if (xg) {
        const float* p = xg + ((size_t)m * 512 + t) * 1024;
        g0 += p[J];       g1 += p[256 + J];
        g2 += p[512 + J]; g3 += p[768 + J];
      }
      float co = c[(size_t)m * 256 + J];
      float cn = sigm(g1) * co + sigm(g0) * tanhf(g2);
      float hh = sigm(g3) * tanhf(cn);
      c[(size_t)m * 256 + J] = cn;
      h_out[(size_t)m * 256 + J] = (bf16)hh;
      if (hstream) hstream[(size_t)m * hs_stride + hs_off + J] = (bf16)hh;
    }
  }
}

// -------------------------------------------------- argmax + embed feedback
__global__ __launch_bounds__(256) void argmax_embed(
    const float* __restrict__ logits, long lstride,
    const float* __restrict__ emb, bf16* __restrict__ inp) {
  int b = blockIdx.x * 8 + (threadIdx.x >> 5);
  int lane = threadIdx.x & 31;
  const float* row = logits + (size_t)b * lstride;
  float best = -3.4e38f;
  int bi = 0;
#pragma unroll
  for (int v = lane; v < 256; v += 32) {
    float val = row[v];
    if (val > best) { best = val; bi = v; }
  }
#pragma unroll
  for (int off = 16; off; off >>= 1) {
    float ob = __shfl_xor(best, off, 32);
    int oi = __shfl_xor(bi, off, 32);
    if (ob > best || (ob == best && oi < bi)) { best = ob; bi = oi; }
  }
  const float* e = emb + (size_t)bi * 256;
#pragma unroll
  for (int k = lane; k < 256; k += 32) {
    inp[(size_t)b * 512 + k] = (bf16)e[k];
    inp[(size_t)b * 512 + 256 + k] = (bf16)0.f;
  }
}

// ---------------------------------------------------------------- launcher
extern "C" void kernel_launch(void* const* d_in, const int* in_sizes, int n_in,
                              void* d_out, int out_size, void* d_ws,
                              size_t ws_size, hipStream_t stream) {
  (void)in_sizes; (void)n_in; (void)out_size; (void)ws_size;
  const int Bb = 256, T = 512, BT = Bb * T, TDEC = 100;

  const int* text = (const int*)d_in[0];
  const float* emb = (const float*)d_in[2];
  const float* e_wih0 = (const float*)d_in[3];
  const float* e_whh0 = (const float*)d_in[4];
  const float* e_b0 = (const float*)d_in[5];
  const float* e_wihr = (const float*)d_in[6];
  const float* e_whhr = (const float*)d_in[7];
  const float* e_br = (const float*)d_in[8];
  const float* d_wih0 = (const float*)d_in[9];
  const float* d_wihr = (const float*)d_in[10];
  const float* d_whh = (const float*)d_in[11];
  const float* d_b = (const float*)d_in[12];
  const float* p_w = (const float*)d_in[13];
  const float* p_b = (const float*)d_in[14];
  float* out = (float*)d_out;

  char* base = (char*)d_ws;
  size_t off = 0;
  auto take = [&](size_t bytes) -> char* {
    char* p = base + off;
    off += (bytes + 255) & ~(size_t)255;
    return p;
  };
  bf16* x_a = (bf16*)take((size_t)BT * 512 * sizeof(bf16));
  bf16* x_b = (bf16*)take((size_t)BT * 512 * sizeof(bf16));
  float* xg = (float*)take((size_t)BT * 1024 * sizeof(float));
  bf16* h_a = (bf16*)take((size_t)Bb * 256 * sizeof(bf16));
  bf16* h_b = (bf16*)take((size_t)Bb * 256 * sizeof(bf16));
  float* c_enc = (float*)take((size_t)Bb * 256 * sizeof(float));
  bf16* dh[3][2];
  float* dc[3];
  for (int l = 0; l < 3; ++l) {
    dh[l][0] = (bf16*)take((size_t)Bb * 256 * sizeof(bf16));
    dh[l][1] = (bf16*)take((size_t)Bb * 256 * sizeof(bf16));
    dc[l] = (float*)take((size_t)Bb * 256 * sizeof(float));
  }
  bf16* inp = (bf16*)take((size_t)Bb * 512 * sizeof(bf16));
  bf16* bw_eih0 = (bf16*)take(2ul * 1024 * 256 * 2);
  bf16* bw_ehh0 = (bf16*)take(2ul * 1024 * 256 * 2);
  bf16* bw_eihr = (bf16*)take(4ul * 1024 * 512 * 2);
  bf16* bw_ehhr = (bf16*)take(4ul * 1024 * 256 * 2);
  bf16* bw_dih0 = (bf16*)take(1024ul * 512 * 2);
  bf16* bw_dihr = (bf16*)take(2ul * 1024 * 256 * 2);
  bf16* bw_dhh = (bf16*)take(3ul * 1024 * 256 * 2);
  bf16* bw_pw = (bf16*)take(256ul * 256 * 2);

  auto cvt = [&](const float* s, bf16* d, size_t n) {
    cvt_f32_bf16<<<(unsigned)(n / 256), 256, 0, stream>>>(s, d, (int)n);
  };
  cvt(e_wih0, bw_eih0, 2ul * 1024 * 256);
  cvt(e_whh0, bw_ehh0, 2ul * 1024 * 256);
  cvt(e_wihr, bw_eihr, 4ul * 1024 * 512);
  cvt(e_whhr, bw_ehhr, 4ul * 1024 * 256);
  cvt(d_wih0, bw_dih0, 1024ul * 512);
  cvt(d_wihr, bw_dihr, 2ul * 1024 * 256);
  cvt(d_whh, bw_dhh, 3ul * 1024 * 256);
  cvt(p_w, bw_pw, 256ul * 256);

  embed_gather<<<(unsigned)(BT * 256 / 256), 256, 0, stream>>>(text, emb, x_a,
                                                               BT * 256);

  // ------------------------------------------------------ encoder (3 biLSTM)
  bf16* xc = x_a;
  bf16* xn = x_b;
  for (int l = 0; l < 3; ++l) {
    int K = (l == 0) ? 256 : 512;
    for (int dir = 0; dir < 2; ++dir) {
      const bf16* wih = (l == 0)
                            ? bw_eih0 + (size_t)dir * 1024 * 256
                            : bw_eihr + (size_t)((l - 1) * 2 + dir) * 1024 * 512;
      const bf16* whh = (l == 0)
                            ? bw_ehh0 + (size_t)dir * 1024 * 256
                            : bw_ehhr + (size_t)((l - 1) * 2 + dir) * 1024 * 256;
      const float* bv =
          (l == 0) ? e_b0 + dir * 1024 : e_br + ((l - 1) * 2 + dir) * 1024;
      // xg = x @ wih^T + b : big parallel WMMA GEMM, 131072 x 1024 x K
      gemm_bias<<<dim3(8, BT / 128), 256, 0, stream>>>(xc, wih, bv, xg, K,
                                                       1024L);
      (void)hipMemsetAsync(h_a, 0, (size_t)Bb * 256 * sizeof(bf16), stream);
      (void)hipMemsetAsync(c_enc, 0, (size_t)Bb * 256 * sizeof(float), stream);
      bf16* hin = h_a;
      bf16* hout = h_b;
      for (int s = 0; s < T; ++s) {
        int t = dir ? (T - 1 - s) : s;
        lstm_step<<<dim3(4, 4), 256, 0, stream>>>(
            nullptr, 0, nullptr, hin, whh, nullptr, xg, t, c_enc, hout, xn,
            (long)T * 512, (long)t * 512 + dir * 256);
        bf16* tmp = hin; hin = hout; hout = tmp;
      }
    }
    bf16* tmp = xc; xc = xn; xn = tmp;
  }
  // xc now holds enc_out (B, T, 512) bf16

  // ---------------------------------------------------------------- decoder
  init_dec_inp<<<(unsigned)(Bb * 512 / 256), 256, 0, stream>>>(xc, inp, T);
  for (int l = 0; l < 3; ++l) {
    (void)hipMemsetAsync(dh[l][0], 0, (size_t)Bb * 256 * sizeof(bf16), stream);
    (void)hipMemsetAsync(dc[l], 0, (size_t)Bb * 256 * sizeof(float), stream);
  }
  int pp[3] = {0, 0, 0};
  for (int t = 0; t < TDEC; ++t) {
    const bf16* xl = inp;
    int xdim = 512;
    const bf16* h_top = nullptr;
    for (int l = 0; l < 3; ++l) {
      const bf16* wih =
          (l == 0) ? bw_dih0 : bw_dihr + (size_t)(l - 1) * 1024 * 256;
      lstm_step<<<dim3(4, 4), 256, 0, stream>>>(
          xl, xdim, wih, dh[l][pp[l]], bw_dhh + (size_t)l * 1024 * 256,
          d_b + l * 1024, nullptr, 0, dc[l], dh[l][1 - pp[l]], nullptr, 0L, 0L);
      xl = dh[l][1 - pp[l]];
      xdim = 256;
      pp[l] = 1 - pp[l];
      h_top = xl;
    }
    // logits = h_top @ proj_w^T + proj_b straight into d_out[:, t, :]
    gemm_bias<<<dim3(2, 2), 256, 0, stream>>>(h_top, bw_pw, p_b,
                                              out + (size_t)t * 256, 256,
                                              (long)TDEC * 256);
    argmax_embed<<<32, 256, 0, stream>>>(out + (size_t)t * 256,
                                         (long)TDEC * 256, emb, inp);
  }
}
